// NovelDistanceLoss_65773129171019
// MI455X (gfx1250) — matrix operand order repaired and checked by reading, not compiled, not run on hardware
//
#include <hip/hip_runtime.h>
#include <hip/hip_bf16.h>

typedef __attribute__((ext_vector_type(16))) __bf16 v16bf;
typedef __attribute__((ext_vector_type(8)))  float  v8f;

#define BZ   16384
#define NR   2048
#define DIM  256
#define MT   64     // rows of wo per workgroup (4 waves x 16)
#define NT   16     // rel rows per n-tile
#define NTILES (NR / NT)
#define EPSV 1e-12f

union Frag {
  v16bf   v;
  __bf16  e[16];
  unsigned u[8];
};

__device__ __forceinline__ v8f wmma_bf16(v16bf a, v16bf b, v8f c) {
  // 8 args: (neg_a, A, neg_b, B, c_mod, C, reuse_a, reuse_b)
  return __builtin_amdgcn_wmma_f32_16x16x32_bf16(false, a, false, b, (short)0, c,
                                                 false, false);
}

// ---------------------------------------------------------------------------
// Kernel R: l2-normalize rel_weight rows, split each f32 into bf16 hi + lo.
// rel stays L2-resident (2 MB) for the whole GEMM afterwards.
// ---------------------------------------------------------------------------
__global__ void __launch_bounds__(256)
rel_norm_split(const float* __restrict__ rel,
               __bf16* __restrict__ relHi,
               __bf16* __restrict__ relLo) {
  __shared__ float red[256];
  __shared__ float sInv;
  const int t = threadIdx.x;
  const int n = blockIdx.x;
  float x = rel[(size_t)n * DIM + t];
  red[t] = x * x;
  __syncthreads();
  for (int s = 128; s > 0; s >>= 1) {
    if (t < s) red[t] += red[t + s];
    __syncthreads();
  }
  if (t == 0) sInv = 1.0f / fmaxf(sqrtf(red[0]), EPSV);
  __syncthreads();
  float xn = x * sInv;
  __bf16 h = (__bf16)xn;
  relHi[(size_t)n * DIM + t] = h;
  relLo[(size_t)n * DIM + t] = (__bf16)(xn - (float)h);
}

// ---------------------------------------------------------------------------
// Kernel C: fused GEMM (bf16-split WMMA, f32 acc) + hard-negative scan.
// B tiles are double-buffered in LDS via GLOBAL_LOAD_ASYNC_TO_LDS_B128
// (ASYNCcnt), so the copy of tile i+1 overlaps the 24 WMMAs of tile i.
// ---------------------------------------------------------------------------
__global__ void __launch_bounds__(128)
hardneg_wmma(const float* __restrict__ wo,
             const int* __restrict__ yv,
             const __bf16* __restrict__ relHi,
             const __bf16* __restrict__ relLo,
             float* __restrict__ partials) {
  __shared__ __bf16 sBh[2][NT * DIM];   // 2 x 8 KB, double buffered
  __shared__ __bf16 sBl[2][NT * DIM];   // 2 x 8 KB
  __shared__ float  sScan[4 * 256];     // one 16x16 f32 tile per wave
  __shared__ float  sRed[128];

  const int  tid  = threadIdx.x;
  const int  wid  = tid >> 5;
  const int  lane = tid & 31;
  const int  lrow = lane & 15;
  const bool hi16 = lane >= 16;
  const int  gm   = blockIdx.x * MT + wid * 16 + lrow;   // my wo row
  const float* wrow = wo + (size_t)gm * DIM;

  // --- row sum-of-squares: lanes L and L^16 each take half the row ---------
  float ss = 0.0f;
  {
    const int kb = hi16 ? 128 : 0;
    #pragma unroll 8
    for (int k = 0; k < 128; ++k) {
      float x = wrow[kb + k];
      ss += x * x;
    }
  }
  ss += __shfl_xor(ss, 16, 32);
  const float invn = 1.0f / fmaxf(sqrtf(ss), EPSV);

  // --- build A fragments in registers (ISA 16-bit A 16x32 layout) ----------
  Frag aHi[8], aLo[8];
  const int khalf = hi16 ? 8 : 0;
  #pragma unroll
  for (int c = 0; c < 8; ++c) {
    const int k0 = c * 32;
    #pragma unroll
    for (int v = 0; v < 8; ++v) {
      const int kk = (v < 4) ? (k0 + khalf + 2 * v)
                             : (k0 + 16 + khalf + 2 * (v - 4));
      float x0 = wrow[kk] * invn;
      float x1 = wrow[kk + 1] * invn;
      __bf16 h0 = (__bf16)x0, h1 = (__bf16)x1;
      aHi[c].e[2 * v]     = h0;
      aHi[c].e[2 * v + 1] = h1;
      aLo[c].e[2 * v]     = (__bf16)(x0 - (float)h0);
      aLo[c].e[2 * v + 1] = (__bf16)(x1 - (float)h1);
    }
  }

  const int myY = yv[gm];
  float posSim  = 1.0f;     // always overwritten: y covers [0, NR)
  float bestSim = -3.0f;    // sims are in [-1, 1]

  float* sc = sScan + wid * 256;
  const int kadd = hi16 ? 8 : 0;

  // async copy of one B tile (hi+lo) into LDS buffer `buf`
  auto prefetch_tile = [&](int n0, int buf) {
    const char* srcH = (const char*)(relHi + (size_t)n0 * DIM);
    const char* srcL = (const char*)(relLo + (size_t)n0 * DIM);
    const unsigned dstH = (unsigned)(uintptr_t)(&sBh[buf][0]);
    const unsigned dstL = (unsigned)(uintptr_t)(&sBl[buf][0]);
    #pragma unroll
    for (int i = 0; i < 4; ++i) {        // 512 x 16B per array / 128 threads
      unsigned off = (unsigned)(tid + i * 128) * 16u;
      asm volatile("global_load_async_to_lds_b128 %0, %1, %2"
                   :: "v"(dstH + off), "v"(off), "s"(srcH) : "memory");
      asm volatile("global_load_async_to_lds_b128 %0, %1, %2"
                   :: "v"(dstL + off), "v"(off), "s"(srcL) : "memory");
    }
  };

  prefetch_tile(0, 0);

  for (int it = 0; it < NTILES; ++it) {
    const int n0  = it * NT;
    const int cur = it & 1;
    // my async portion of tile `it` landed:
    asm volatile("s_wait_asynccnt 0x0" ::: "memory");
    // everyone's portion landed; prior reads of the other buffer finished:
    __syncthreads();
    if (it + 1 < NTILES) prefetch_tile(n0 + NT, cur ^ 1);

    const unsigned* bhu = (const unsigned*)&sBh[cur][0];
    const unsigned* blu = (const unsigned*)&sBl[cur][0];

    v8f acc0 = {}, acc1 = {}, acc2 = {};   // 3 independent WMMA chains
    #pragma unroll
    for (int c = 0; c < 8; ++c) {
      // B fragment (ISA 16-bit B 32x16 layout): lane = N, u32 = K pair
      Frag bh, bl;
      const int base = (lrow * DIM + c * 32 + kadd) >> 1;
      #pragma unroll
      for (int v = 0; v < 8; ++v) {
        bh.u[v] = bhu[base + v + (v & 4)];
        bl.u[v] = blu[base + v + (v & 4)];
      }
      acc0 = wmma_bf16(aHi[c].v, bh.v, acc0);  // hi*hi
      acc1 = wmma_bf16(aHi[c].v, bl.v, acc1);  // hi*lo
      acc2 = wmma_bf16(aLo[c].v, bh.v, acc2);  // lo*hi
    }

    // spill 16x16 sim tile to this wave's LDS area, then per-row scan
    #pragma unroll
    for (int r = 0; r < 8; ++r)
      sc[(r + (hi16 ? 8 : 0)) * 16 + lrow] = acc0[r] + acc1[r] + acc2[r];

    if (lane < 16) {
      #pragma unroll
      for (int n = 0; n < 16; ++n) {
        float vs = sc[lane * 16 + n];
        int ng = n0 + n;
        if (ng == myY) posSim = vs;                  // positive class excluded
        else           bestSim = fmaxf(bestSim, vs); // hardest negative
      }
    }
  }

  // pos/neg distances from sims: ||a-b|| = sqrt(clip(2 - 2 a.b)) for unit vecs
  float contrib = 0.0f;
  if (lane < 16) {
    float pd = sqrtf(fmaxf(2.0f - 2.0f * posSim, 0.0f));
    float nd = sqrtf(fmaxf(2.0f - 2.0f * bestSim, 0.0f));
    contrib = pd + 1.0f - nd;   // margin = 1.0
  }
  sRed[tid] = contrib;
  __syncthreads();
  for (int s = 64; s > 0; s >>= 1) {
    if (tid < s) sRed[tid] += sRed[tid + s];
    __syncthreads();
  }
  if (tid == 0) partials[blockIdx.x] = sRed[0];
}

// ---------------------------------------------------------------------------
// Kernel D: deterministic final reduction of the 256 workgroup partials.
// ---------------------------------------------------------------------------
__global__ void __launch_bounds__(256)
final_reduce(const float* __restrict__ partials, float* __restrict__ out) {
  __shared__ float red[256];
  const int t = threadIdx.x;
  red[t] = partials[t];
  __syncthreads();
  for (int s = 128; s > 0; s >>= 1) {
    if (t < s) red[t] += red[t + s];
    __syncthreads();
  }
  if (t == 0) out[0] = red[0] * (1.0f / (float)BZ);
}

extern "C" void kernel_launch(void* const* d_in, const int* in_sizes, int n_in,
                              void* d_out, int out_size, void* d_ws, size_t ws_size,
                              hipStream_t stream) {
  (void)in_sizes; (void)n_in; (void)out_size; (void)ws_size;
  const float* wo   = (const float*)d_in[0];
  const float* rel  = (const float*)d_in[1];
  const int*   in_y = (const int*)d_in[2];

  char* w = (char*)d_ws;
  __bf16* relHi    = (__bf16*)w;                                        // 1 MB
  __bf16* relLo    = (__bf16*)(w + (size_t)NR * DIM * sizeof(__bf16));  // 1 MB
  float*  partials = (float*)(w + 2 * (size_t)NR * DIM * sizeof(__bf16));
  float*  out      = (float*)d_out;

  rel_norm_split<<<NR, 256, 0, stream>>>(rel, relHi, relLo);
  hardneg_wmma<<<BZ / MT, 128, 0, stream>>>(wo, in_y, relHi, relLo, partials);
  final_reduce<<<1, 256, 0, stream>>>(partials, out);
}
